// GATNet_1486058684430
// MI455X (gfx1250) — matrix operand order repaired and checked by reading, not compiled, not run on hardware
//
#include <hip/hip_runtime.h>
#include <hip/hip_bf16.h>

// ---------------------------------------------------------------------------
// GATv2 network for MI455X (gfx1250, wave32).
//  - GEMMs (x@Wl, x@Wr) on v_wmma_f32_16x16x32_bf16 (bf16 A/B, f32 acc).
//    Register-blocked: each wave owns a 32x64 output tile (2x4 WMMA tiles),
//    so each 32-deep K step issues 8 WMMAs against 12 b128 loads instead of
//    1 WMMA : 4 loads.  Activations (<=25.7MB bf16) live in the 192MB L2,
//    so direct global 16B loads per lane (matching the ISA A/B layouts)
//    are sufficient -- no LDS/TDM staging needed at ~28 GFLOP total.
//  - Edge phase (850K edges, softmax-by-dst, scatter-add) is the bandwidth-
//    bound part -> f32 VALU kernels with global f32 atomics; segment-max via
//    order-preserving uint encoding + atomicMax(u32).
// ---------------------------------------------------------------------------

typedef __attribute__((ext_vector_type(16))) __bf16 v16bf;
typedef __attribute__((ext_vector_type(8)))  __bf16 v8bf;
typedef __attribute__((ext_vector_type(8)))  float  v8f;

#define NNODES   50000
#define NPAD     50048          // multiple of 128 (block M tile)
#define NEDGE_IN 800000
#define NEDGE    850000         // + 50000 self loops
#define ENC_NEG_INF 0x007FFFFFu // enc(-inf) under order-preserving map

// ---- helpers ---------------------------------------------------------------

__device__ __forceinline__ void edge_sd(const int* __restrict__ ei, int e,
                                        int& s, int& d) {
  if (e < NEDGE_IN) { s = ei[e]; d = ei[NEDGE_IN + e]; }
  else              { s = d = e - NEDGE_IN; }      // appended self loop
}

__device__ __forceinline__ unsigned f2ord(float f) {
  unsigned b = __float_as_uint(f);
  return (b & 0x80000000u) ? ~b : (b | 0x80000000u);
}
__device__ __forceinline__ float ord2f(unsigned e) {
  unsigned b = (e & 0x80000000u) ? (e & 0x7FFFFFFFu) : ~e;
  return __uint_as_float(b);
}

// ---- conversion / setup kernels -------------------------------------------

// x[NNODES,2] fp32 -> Xb[NPAD,32] bf16 (zero padded rows + K)
__global__ void cvt_x0(const float* __restrict__ x, __bf16* __restrict__ xb) {
  int i = blockIdx.x * blockDim.x + threadIdx.x;
  if (i >= NPAD * 32) return;
  int n = i >> 5, k = i & 31;
  float v = (n < NNODES && k < 2) ? x[n * 2 + k] : 0.f;
  xb[i] = (__bf16)v;
}

// W[K,D] fp32 -> Wt[Dw,Kp] bf16 (transposed, zero padded to Dw x Kp)
__global__ void cvt_wt(const float* __restrict__ W, __bf16* __restrict__ Wt,
                       int K, int D, int Kp, int Dw) {
  int i = blockIdx.x * blockDim.x + threadIdx.x;
  if (i >= Dw * Kp) return;
  int n = i / Kp, k = i - n * Kp;
  float v = (n < D && k < K) ? W[(size_t)k * D + n] : 0.f;
  Wt[i] = (__bf16)v;
}

__global__ void init_layer(float* __restrict__ outw, float* __restrict__ denom,
                           unsigned* __restrict__ emaxE, int cntOut, int cntNH) {
  int i = blockIdx.x * blockDim.x + threadIdx.x;
  if (i < cntOut) outw[i] = 0.f;
  if (i < cntNH) { denom[i] = 0.f; emaxE[i] = ENC_NEG_INF; }
}

// ---- WMMA GEMM: Y[NPAD,Dw] = Xb[NPAD,Kp] @ Wt^T  (Wt is [Dw,Kp]) ----------
// Register-blocked: each wave computes a 32x64 tile (2 M x 4 N 16x16 tiles).
// 8 waves/block arranged 4(M) x 2(N) -> 128x128 block tile.
__global__ void __launch_bounds__(256)
wmma_gemm_bf16(const __bf16* __restrict__ Xb, const __bf16* __restrict__ Wt,
               float* __restrict__ Y, int Kp, int Dw) {
  const int wave = threadIdx.x >> 5;
  const int lane = threadIdx.x & 31;
  const int half = lane >> 4;     // which 16-lane half of the wave
  const int l16  = lane & 15;
  const int rowBase = blockIdx.x * 128 + (wave >> 1) * 32;
  const int colBase = blockIdx.y * 128 + (wave & 1) * 64;
  if (colBase >= Dw) return;      // wave-uniform: EXEC stays all-1s for WMMA

  v8f acc[2][4] = {};
  const __bf16* arow0 = Xb + (size_t)(rowBase + l16) * Kp;        // M 0..15
  const __bf16* arow1 = arow0 + (size_t)16 * Kp;                  // M 16..31
  const __bf16* brow0 = Wt + (size_t)(colBase + l16) * Kp;        // N j*16..

  for (int kb = 0; kb < Kp; kb += 32) {
    // A 16x32 bf16: lanes 0-15 (M=0..15) hold K 0-7 (v0-3) & 16-23 (v4-7);
    //               lanes 16-31 hold K 8-15 & 24-31  -> two 16B loads each.
    v16bf a0, a1;
    *((v8bf*)&a0)       = *(const v8bf*)(arow0 + kb + half * 8);
    *(((v8bf*)&a0) + 1) = *(const v8bf*)(arow0 + kb + 16 + half * 8);
    *((v8bf*)&a1)       = *(const v8bf*)(arow1 + kb + half * 8);
    *(((v8bf*)&a1) + 1) = *(const v8bf*)(arow1 + kb + 16 + half * 8);
#pragma unroll
    for (int j = 0; j < 4; ++j) {
      // B 32x16 bf16: lanes 0-15 hold K 0-7 & 16-23, lanes 16-31 hold K 8-15 & 24-31;
      // column n per lane -> Wt row is contiguous K -> two 16B loads.
      const __bf16* br = brow0 + (size_t)(j * 16) * Kp;
      v16bf b;
      *((v8bf*)&b)       = *(const v8bf*)(br + kb + half * 8);
      *(((v8bf*)&b) + 1) = *(const v8bf*)(br + kb + 16 + half * 8);
      acc[0][j] = __builtin_amdgcn_wmma_f32_16x16x32_bf16(
          false, a0, false, b, (short)0, acc[0][j], false, false);
      acc[1][j] = __builtin_amdgcn_wmma_f32_16x16x32_bf16(
          false, a1, false, b, (short)0, acc[1][j], false, false);
    }
  }
  // C/D f32 16x16: vgpr r -> M = r + half*8, N = l16
#pragma unroll
  for (int mi = 0; mi < 2; ++mi) {
#pragma unroll
    for (int j = 0; j < 4; ++j) {
      float* yb = Y + (size_t)(rowBase + mi * 16) * Dw + colBase + j * 16;
#pragma unroll
      for (int r = 0; r < 8; ++r)
        yb[(size_t)(r + half * 8) * Dw + l16] = acc[mi][j][r];
    }
  }
}

// ---- edge-phase kernels ----------------------------------------------------

// e[e,h] = sum_c leakyrelu(xl[src,h,c]+xr[dst,h,c]) * att[h,c]; segment max
__global__ void score_k(const float* __restrict__ xl, const float* __restrict__ xr,
                        const float* __restrict__ att, const int* __restrict__ ei,
                        float* __restrict__ esc, unsigned* __restrict__ emaxE,
                        int H, int C, int Dw) {
  int id = blockIdx.x * blockDim.x + threadIdx.x;
  if (id >= NEDGE * H) return;
  int e = id / H, h = id - e * H;
  int s, d; edge_sd(ei, e, s, d);
  const float* pl = xl + (size_t)s * Dw + h * C;
  const float* pr = xr + (size_t)d * Dw + h * C;
  const float* pa = att + h * C;
  float acc = 0.f;
  for (int c = 0; c < C; ++c) {
    float v = pl[c] + pr[c];
    v = v > 0.f ? v : 0.2f * v;
    acc = fmaf(v, pa[c], acc);
  }
  esc[id] = acc;
  atomicMax(&emaxE[(size_t)d * H + h], f2ord(acc));
}

__global__ void exp_k(float* __restrict__ esc, const unsigned* __restrict__ emaxE,
                      float* __restrict__ denom, const int* __restrict__ ei, int H) {
  int id = blockIdx.x * blockDim.x + threadIdx.x;
  if (id >= NEDGE * H) return;
  int e = id / H, h = id - e * H;
  int s, d; edge_sd(ei, e, s, d);
  (void)s;
  float m  = ord2f(emaxE[(size_t)d * H + h]);
  float ex = __expf(esc[id] - m);
  esc[id] = ex;
  atomicAdd(&denom[(size_t)d * H + h], ex);
}

__global__ void agg_k(const float* __restrict__ esc, const float* __restrict__ denom,
                      const float* __restrict__ xl, const int* __restrict__ ei,
                      float* __restrict__ outw, int H, int C, int Dw) {
  int id = blockIdx.x * blockDim.x + threadIdx.x;
  if (id >= NEDGE * H) return;
  int e = id / H, h = id - e * H;
  int s, d; edge_sd(ei, e, s, d);
  float alpha = esc[id] / (denom[(size_t)d * H + h] + 1e-16f);
  const float* pl = xl + (size_t)s * Dw + h * C;
  float* po = outw + (size_t)d * Dw + h * C;
  for (int c = 0; c < C; ++c) atomicAdd(&po[c], alpha * pl[c]);
}

// concat path: y = out + b, ELU, write bf16 input for next layer (stride KpNext)
__global__ void fin_concat(const float* __restrict__ outw, const float* __restrict__ b,
                           __bf16* __restrict__ xnext, int D, int Dw, int KpNext) {
  int i = blockIdx.x * blockDim.x + threadIdx.x;
  if (i >= NPAD * KpNext) return;
  int n = i / KpNext, c = i - n * KpNext;
  float v = 0.f;
  if (n < NNODES && c < D) {
    v = outw[(size_t)n * Dw + c] + b[c];
    v = v > 0.f ? v : (__expf(v) - 1.f);         // ELU (all but last layer)
  }
  xnext[i] = (__bf16)v;
}

// last layer: mean over heads + bias, fp32 out
__global__ void fin_mean(const float* __restrict__ outw, const float* __restrict__ b,
                         float* __restrict__ y, int H, int C, int Dw) {
  int i = blockIdx.x * blockDim.x + threadIdx.x;
  if (i >= NNODES * C) return;
  int n = i / C, c = i - n * C;
  float s = 0.f;
  for (int h = 0; h < H; ++h) s += outw[(size_t)n * Dw + h * C + c];
  y[i] = s / (float)H + b[c];
}

// ---- host orchestration ----------------------------------------------------

extern "C" void kernel_launch(void* const* d_in, const int* in_sizes, int n_in,
                              void* d_out, int out_size, void* d_ws, size_t ws_size,
                              hipStream_t stream) {
  (void)in_sizes; (void)n_in; (void)out_size; (void)ws_size;
  const float* x  = (const float*)d_in[0];
  const int*   ei = (const int*)d_in[1];

  static const int CIN[6]  = {2, 128, 256, 256, 128, 64};
  static const int COUT[6] = {16, 32, 32, 16, 16, 2};
  static const int HD[6]   = {8, 8, 8, 8, 4, 1};

  auto cdiv = [](int a, int b) { return (a + b - 1) / b; };

  // workspace partition (256B aligned)
  char* p = (char*)d_ws;
  auto take = [&](size_t bytes) -> void* {
    void* r = p; p += (bytes + 255) & ~(size_t)255; return r;
  };
  __bf16*   Xb    = (__bf16*)take((size_t)NPAD * 256 * 2);
  __bf16*   WtL   = (__bf16*)take((size_t)256 * 256 * 2);
  __bf16*   WtR   = (__bf16*)take((size_t)256 * 256 * 2);
  float*    xl    = (float*)take((size_t)NPAD * 256 * 4);
  float*    xr    = (float*)take((size_t)NPAD * 256 * 4);
  float*    outw  = (float*)take((size_t)NPAD * 256 * 4);
  float*    esc   = (float*)take((size_t)NEDGE * 8 * 4);
  unsigned* emaxE = (unsigned*)take((size_t)NPAD * 8 * 4);
  float*    denom = (float*)take((size_t)NPAD * 8 * 4);

  cvt_x0<<<cdiv(NPAD * 32, 256), 256, 0, stream>>>(x, Xb);

  int Kp = 32;  // layer-0 cin=2 padded to 32
  for (int li = 0; li < 6; ++li) {
    const int cin = CIN[li], C = COUT[li], H = HD[li];
    const int D = H * C;
    const int Dw = (D + 63) & ~63;      // pad N to wave tile width (64)
    const float* Wl  = (const float*)d_in[2 + 4 * li + 0];
    const float* Wr  = (const float*)d_in[2 + 4 * li + 1];
    const float* att = (const float*)d_in[2 + 4 * li + 2];
    const float* bia = (const float*)d_in[2 + 4 * li + 3];

    cvt_wt<<<cdiv(Dw * Kp, 256), 256, 0, stream>>>(Wl, WtL, cin, D, Kp, Dw);
    cvt_wt<<<cdiv(Dw * Kp, 256), 256, 0, stream>>>(Wr, WtR, cin, D, Kp, Dw);

    dim3 gg(NPAD / 128, cdiv(Dw, 128));
    wmma_gemm_bf16<<<gg, 256, 0, stream>>>(Xb, WtL, xl, Kp, Dw);
    wmma_gemm_bf16<<<gg, 256, 0, stream>>>(Xb, WtR, xr, Kp, Dw);

    const int cntOut = NPAD * Dw, cntNH = NPAD * H;
    const int cntMax = cntOut > cntNH ? cntOut : cntNH;
    init_layer<<<cdiv(cntMax, 256), 256, 0, stream>>>(outw, denom, emaxE, cntOut, cntNH);

    const int EH = NEDGE * H;
    score_k<<<cdiv(EH, 256), 256, 0, stream>>>(xl, xr, att, ei, esc, emaxE, H, C, Dw);
    exp_k  <<<cdiv(EH, 256), 256, 0, stream>>>(esc, emaxE, denom, ei, H);
    agg_k  <<<cdiv(EH, 256), 256, 0, stream>>>(esc, denom, xl, ei, outw, H, C, Dw);

    if (li < 5) {
      const int KpNext = D;  // next layer cin == D, always a multiple of 32
      fin_concat<<<cdiv(NPAD * KpNext, 256), 256, 0, stream>>>(outw, bia, Xb, D, Dw, KpNext);
      Kp = KpNext;
    } else {
      fin_mean<<<cdiv(NNODES * C, 256), 256, 0, stream>>>(outw, bia, (float*)d_out, H, C, Dw);
    }
  }
}
